// ProbAttention_43396349559213
// MI455X (gfx1250) — matrix operand order repaired and checked by reading, not compiled, not run on hardware
//
#include <hip/hip_runtime.h>
#include <hip/hip_bf16.h>
#include <math.h>

typedef __attribute__((ext_vector_type(16))) _Float16 v16h;
typedef __attribute__((ext_vector_type(8)))  _Float16 v8h;
typedef __attribute__((ext_vector_type(4)))  _Float16 v4h;
typedef __attribute__((ext_vector_type(8)))  float    v8f;
typedef __attribute__((ext_vector_type(4)))  float    v4f;

union V16U { v16h v; v8h h[2]; };

#define QPITCH 72   // qs row pitch (halfs): 144B, 16B-aligned
#define KPITCH 72   // ks row pitch (halfs)
#define VTP    40   // vT row pitch (halfs): 80B, 16B-aligned

// ---------------------------------------------------------------------------
// Phase 1: sparsity measure M[b,h,q] = max_s(q.k_s) - sum_s(q.k_s)/L
// One thread per (b,h,q); irregular per-query K gather -> VALU f32 dots.
// ---------------------------------------------------------------------------
__global__ void score_kernel(const float* __restrict__ q,
                             const float* __restrict__ k,
                             const int*   __restrict__ sample_idx,
                             float* __restrict__ M,
                             int B, int L, int H, int D, int SK) {
    int gid = blockIdx.x * blockDim.x + threadIdx.x;
    int total = B * H * L;
    if (gid >= total) return;
    int qpos = gid % L;
    int h    = (gid / L) % H;
    int b    = gid / (L * H);

    const float* qrow = q + ((size_t)((size_t)b * L + qpos) * H + h) * D;
    float qv[64];
#pragma unroll
    for (int d0 = 0; d0 < 64; d0 += 4) {
        v4f t = *(const v4f*)(qrow + d0);
        qv[d0] = t.x; qv[d0+1] = t.y; qv[d0+2] = t.z; qv[d0+3] = t.w;
    }

    float mx = -INFINITY, sm = 0.0f;
    for (int s = 0; s < SK; ++s) {
        int idx = sample_idx[qpos * SK + s];
        const float* krow = k + ((size_t)((size_t)b * L + idx) * H + h) * D;
        float acc = 0.0f;
#pragma unroll
        for (int d0 = 0; d0 < 64; d0 += 4) {
            v4f t = *(const v4f*)(krow + d0);
            acc = fmaf(qv[d0],   t.x, acc);
            acc = fmaf(qv[d0+1], t.y, acc);
            acc = fmaf(qv[d0+2], t.z, acc);
            acc = fmaf(qv[d0+3], t.w, acc);
        }
        mx = fmaxf(mx, acc);
        sm += acc;
    }
    M[gid] = mx - sm / (float)L;
}

// ---------------------------------------------------------------------------
// Phase 2: top-U indices per (b,h) row of M. Iterative argmax with
// lowest-index tie break == jax.lax.top_k ordering.
// ---------------------------------------------------------------------------
__global__ void topk_kernel(const float* __restrict__ M,
                            int* __restrict__ top, int L, int U) {
    __shared__ float sv[2048];
    __shared__ float rv[256];
    __shared__ int   ri[256];
    int bh = blockIdx.x;
    const float* Mrow = M + (size_t)bh * L;
    for (int i = threadIdx.x; i < L; i += 256) sv[i] = Mrow[i];
    __syncthreads();

    for (int it = 0; it < U; ++it) {
        float best = -INFINITY; int bi = L;
        for (int i = threadIdx.x; i < L; i += 256) {
            float v = sv[i];
            if (v > best || (v == best && i < bi)) { best = v; bi = i; }
        }
        rv[threadIdx.x] = best; ri[threadIdx.x] = bi;
        __syncthreads();
        for (int off = 128; off > 0; off >>= 1) {
            if ((int)threadIdx.x < off) {
                float v2 = rv[threadIdx.x + off]; int i2 = ri[threadIdx.x + off];
                if (v2 > rv[threadIdx.x] ||
                    (v2 == rv[threadIdx.x] && i2 < ri[threadIdx.x])) {
                    rv[threadIdx.x] = v2; ri[threadIdx.x] = i2;
                }
            }
            __syncthreads();
        }
        if (threadIdx.x == 0) { top[bh * U + it] = ri[0]; sv[ri[0]] = -INFINITY; }
        __syncthreads();
    }
}

// ---------------------------------------------------------------------------
// Phase 3: out = v. Dominant HBM cost (32MB each way). Non-temporal stores
// keep the output stream from evicting Q/K/V, which phase 4 re-reads from L2.
// ---------------------------------------------------------------------------
__global__ void copy_v_kernel(const v4f* __restrict__ v,
                              v4f* __restrict__ out, int n4) {
    int i = blockIdx.x * blockDim.x + threadIdx.x;
    if (i < n4) {
        v4f t = v[i];
        __builtin_nontemporal_store(t, &out[i]);
    }
}

// ---------------------------------------------------------------------------
// Phase 4: flash-attention for the selected queries, WMMA f16->f32.
// One 128-thread block per (b,h): waves 0..2 each own a 16-query tile and
// share one staged K/V chunk in LDS; wave 3 only helps stage.
// All WMMA fragments are contiguous 16B LDS loads (V stored transposed).
// ---------------------------------------------------------------------------
__global__ void __launch_bounds__(128)
sparse_attn_kernel(const float* __restrict__ q,
                   const float* __restrict__ k,
                   const float* __restrict__ v,
                   const int*   __restrict__ top,
                   float* __restrict__ out,
                   int B, int L, int H, int D, int U) {
    const int NT = (U + 15) / 16;      // 3 query tiles of 16 (rows >= U zero-padded)
    int bh   = blockIdx.x;
    int h    = bh % H;
    int b    = bh / H;
    int tid  = threadIdx.x;
    int t    = tid >> 5;               // wave id: 0..2 compute, 3 staging-only
    int lane = tid & 31;
    int m16  = lane & 15;
    int half = lane >> 4;

    __shared__ __align__(16) _Float16 qs[3][16][QPITCH];
    __shared__ __align__(16) _Float16 ks[32][KPITCH];
    __shared__ __align__(16) _Float16 vT[64][VTP];       // vT[d][c] = V[c][d]
    __shared__ __align__(16) float    sS[3][16][32];
    __shared__ __align__(16) _Float16 sP[3][16][32];
    __shared__ float row_m[3][16], row_l[3][16], row_c[3][16];

    const size_t rowstride = (size_t)H * D;              // floats between L rows
    const size_t bhbase    = ((size_t)b * L * H + h) * D;

    // ---- stage q_red: 48 rows x 64 halfs (zero pad rows >= U) ----
    for (int e = tid; e < 48 * 64; e += 128) {
        int r = e >> 6;            // selected-query row 0..47
        int d = e & 63;
        float val = 0.0f;
        if (r < U) {
            int idx = top[bh * U + r];
            val = q[bhbase + (size_t)idx * rowstride + d];
        }
        qs[r >> 4][r & 15][d] = (_Float16)val;
    }
    if (t < NT && lane < 16) { row_m[t][lane] = -INFINITY; row_l[t][lane] = 0.0f; }
    __syncthreads();

    // ---- per-wave Q A-fragments (16x32 layout), contiguous 16B LDS loads ----
    V16U aq0, aq1;
    {
        const _Float16* qr = &qs[t < NT ? t : 0][m16][0];
        aq0.h[0] = *(const v8h*)(qr + 16 * half);
        aq0.h[1] = *(const v8h*)(qr + 8 + 16 * half);
        aq1.h[0] = *(const v8h*)(qr + 32 + 16 * half);
        aq1.h[1] = *(const v8h*)(qr + 40 + 16 * half);
    }

    v8f o[4] = { {}, {}, {}, {} };
    const float scale = 0.125f;    // 1/sqrt(64)

    // staging role: each thread owns row = tid/4, d-slice = (tid%4)*16
    const int srow  = tid >> 2;
    const int dbase = (tid & 3) << 4;
    const float* kstage = k + bhbase + (size_t)srow * rowstride + dbase;
    const float* vstage = v + bhbase + (size_t)srow * rowstride + dbase;
    const size_t chunkstep = 32 * rowstride;

    for (int c0 = 0; c0 < L; c0 += 32) {
        // ---- stage K (row-major f16) and V (transposed f16) ----
        {
            const float* kr = kstage + (size_t)c0 * rowstride;
            const float* vr = vstage + (size_t)c0 * rowstride;
#pragma unroll
            for (int i = 0; i < 4; ++i) {
                v4f f = *(const v4f*)(kr + i * 4);
                v4h hk; hk[0] = (_Float16)f.x; hk[1] = (_Float16)f.y;
                hk[2] = (_Float16)f.z; hk[3] = (_Float16)f.w;
                *(v4h*)&ks[srow][dbase + i * 4] = hk;

                v4f g = *(const v4f*)(vr + i * 4);
                vT[dbase + i * 4 + 0][srow] = (_Float16)g.x;
                vT[dbase + i * 4 + 1][srow] = (_Float16)g.y;
                vT[dbase + i * 4 + 2][srow] = (_Float16)g.z;
                vT[dbase + i * 4 + 3][srow] = (_Float16)g.w;
            }
            if (c0 + 32 < L) {   // prefetch next chunk -> global_prefetch_b8
                __builtin_prefetch(kr + chunkstep, 0, 1);
                __builtin_prefetch(vr + chunkstep, 0, 1);
            }
        }
        __syncthreads();

        if (t < NT) {
            // ---- S = Q @ K^T : 2 column tiles, K=64 = 2 WMMAs each ----
#pragma unroll
            for (int nt = 0; nt < 2; ++nt) {
                v8f c8 = {};
                const _Float16* kr = &ks[nt * 16 + m16][16 * half];
                V16U bf;
                bf.h[0] = *(const v8h*)(kr);
                bf.h[1] = *(const v8h*)(kr + 8);
                c8 = __builtin_amdgcn_wmma_f32_16x16x32_f16(false, aq0.v, false, bf.v,
                                                            (short)0, c8, false, false);
                bf.h[0] = *(const v8h*)(kr + 32);
                bf.h[1] = *(const v8h*)(kr + 40);
                c8 = __builtin_amdgcn_wmma_f32_16x16x32_f16(false, aq1.v, false, bf.v,
                                                            (short)0, c8, false, false);
#pragma unroll
                for (int j = 0; j < 8; ++j)
                    sS[t][j + 8 * half][nt * 16 + m16] = c8[j] * scale;
            }

            // ---- online softmax over this 32-key chunk (lane r<16 owns row r) ----
            if (lane < 16) {
                int r = lane;
                float mold = row_m[t][r];
                float mc = mold;
#pragma unroll
                for (int c = 0; c < 32; ++c) mc = fmaxf(mc, sS[t][r][c]);
                float corr = __expf(mold - mc);   // 0 when mold == -inf
                float sum = 0.0f;
#pragma unroll
                for (int c = 0; c < 32; ++c) {
                    float e = __expf(sS[t][r][c] - mc);
                    sP[t][r][c] = (_Float16)e;
                    sum += e;
                }
                row_l[t][r] = row_l[t][r] * corr + sum;
                row_m[t][r] = mc;
                row_c[t][r] = corr;
            }

            // ---- rescale O, then O += P @ V (4 d-tiles, 1 WMMA each) ----
#pragma unroll
            for (int j = 0; j < 8; ++j) {
                float corr = row_c[t][j + 8 * half];
                o[0][j] *= corr; o[1][j] *= corr; o[2][j] *= corr; o[3][j] *= corr;
            }
            V16U ap;
            {
                const _Float16* pr = &sP[t][m16][0];
                ap.h[0] = *(const v8h*)(pr + 16 * half);
                ap.h[1] = *(const v8h*)(pr + 8 + 16 * half);
            }
#pragma unroll
            for (int dt = 0; dt < 4; ++dt) {
                const _Float16* vr = &vT[dt * 16 + m16][16 * half];
                V16U bv;
                bv.h[0] = *(const v8h*)(vr);
                bv.h[1] = *(const v8h*)(vr + 8);
                o[dt] = __builtin_amdgcn_wmma_f32_16x16x32_f16(false, ap.v, false, bv.v,
                                                               (short)0, o[dt], false, false);
            }
        }
        __syncthreads();
    }

    // ---- epilogue: normalize and scatter ctx rows into out (out == v already) ----
    if (t < NT) {
#pragma unroll
        for (int j = 0; j < 8; ++j) {
            int m = j + 8 * half;
            int u = t * 16 + m;
            if (u < U) {
                int idx = top[bh * U + u];
                float inv = 1.0f / row_l[t][m];
                float* orow = out + bhbase + (size_t)idx * rowstride;
                orow[ 0 + m16] = o[0][j] * inv;
                orow[16 + m16] = o[1][j] * inv;
                orow[32 + m16] = o[2][j] * inv;
                orow[48 + m16] = o[3][j] * inv;
            }
        }
    }
}

// ---------------------------------------------------------------------------
extern "C" void kernel_launch(void* const* d_in, const int* in_sizes, int n_in,
                              void* d_out, int out_size, void* d_ws, size_t ws_size,
                              hipStream_t stream) {
    const float* q = (const float*)d_in[0];
    const float* k = (const float*)d_in[1];
    const float* v = (const float*)d_in[2];
    const int*   sample_idx = (const int*)d_in[3];
    // d_in[4] = attn_mask scalar, unused (==0)

    const int B = 4, L = 2048, H = 16, D = 64;
    const int SK = in_sizes[3] / L;       // 40
    const int U  = SK;                    // u_top == sample_k here

    float* M   = (float*)d_ws;
    int*   top = (int*)((char*)d_ws + (size_t)B * H * L * sizeof(float));
    float* out = (float*)d_out;

    int total = B * H * L;
    hipLaunchKernelGGL(score_kernel, dim3((total + 255) / 256), dim3(256), 0, stream,
                       q, k, sample_idx, M, B, L, H, D, SK);

    hipLaunchKernelGGL(topk_kernel, dim3(B * H), dim3(256), 0, stream, M, top, L, U);

    int n4 = (B * L * H * D) / 4;
    hipLaunchKernelGGL(copy_v_kernel, dim3((n4 + 255) / 256), dim3(256), 0, stream,
                       (const v4f*)v, (v4f*)out, n4);

    hipLaunchKernelGGL(sparse_attn_kernel, dim3(B * H), dim3(128), 0, stream,
                       q, k, v, top, out, B, L, H, D, U);
}
